// HighOrderAttention_58256936403216
// MI455X (gfx1250) — compile-verified, matrix operand order checked by reading, not run on hardware
//
#include <hip/hip_runtime.h>
#include <hip/hip_bf16.h>


#define BB   2
#define NN   192
#define DIMM 512
#define HH   8
#define DH   64
#define SCALE 0.125f

typedef _Float16 half_t;
typedef __attribute__((ext_vector_type(16))) _Float16 v16h;
typedef __attribute__((ext_vector_type(8)))  _Float16 v8h;
typedef __attribute__((ext_vector_type(8)))  float    v8f;
typedef __attribute__((ext_vector_type(4)))  float    v4f;
typedef __attribute__((ext_vector_type(16))) int      v16i;
typedef __attribute__((ext_vector_type(8)))  int      v8i;
typedef __attribute__((ext_vector_type(2)))  int      v2i;
typedef __attribute__((address_space(3)))    half_t   lds_half;

#define KS  72
#define VT  200
#define PS  200
#define PS8 208

__device__ __forceinline__ v16h load_frag_h(const half_t* X, int row, int ld,
                                            int c0, int hi) {
    const half_t* base = X + row * ld + c0 + hi * 8;
    v8h lo = *(const v8h*)(base);
    v8h hh = *(const v8h*)(base + 16);
    v16h a;
#pragma unroll
    for (int i = 0; i < 8; ++i) { a[i] = lo[i]; a[8 + i] = hh[i]; }
    return a;
}

__device__ __forceinline__ v16h load_frag_f32(const float* X, int row, int ld,
                                              int c0, int hi) {
    const float* base = X + (size_t)row * ld + c0 + hi * 8;
    v4f f0 = *(const v4f*)(base);
    v4f f1 = *(const v4f*)(base + 4);
    v4f f2 = *(const v4f*)(base + 16);
    v4f f3 = *(const v4f*)(base + 20);
    v16h a;
#pragma unroll
    for (int i = 0; i < 4; ++i) {
        a[i]      = (half_t)f0[i];
        a[4 + i]  = (half_t)f1[i];
        a[8 + i]  = (half_t)f2[i];
        a[12 + i] = (half_t)f3[i];
    }
    return a;
}

__device__ __forceinline__ v16i load_frag8_128(const unsigned char* X, int row,
                                               int ld, int c0, int hi) {
    const unsigned char* base = X + row * ld + c0 + hi * 8;
    v16i a;
#pragma unroll
    for (int r = 0; r < 8; ++r) {
        v2i w = *(const v2i*)(base + r * 16);
        a[2 * r] = w[0]; a[2 * r + 1] = w[1];
    }
    return a;
}

__device__ __forceinline__ v8i load_frag8_64(const unsigned char* X, int row,
                                             int ld, int c0, int hi) {
    const unsigned char* base = X + row * ld + c0 + hi * 8;
    v8i a;
#pragma unroll
    for (int r = 0; r < 4; ++r) {
        v2i w = *(const v2i*)(base + r * 16);
        a[2 * r] = w[0]; a[2 * r + 1] = w[1];
    }
    return a;
}

__device__ __forceinline__ v8f wmma_f16(v16h a, v16h b, v8f c) {
    return __builtin_amdgcn_wmma_f32_16x16x32_f16(false, a, false, b,
                                                  (short)0, c, false, false);
}

__global__ __launch_bounds__(128) void proj_kernel(
    const float* __restrict__ queries, const float* __restrict__ keys,
    const float* __restrict__ values,
    const float* __restrict__ Wq, const float* __restrict__ bq,
    const float* __restrict__ Wk, const float* __restrict__ bk,
    const float* __restrict__ Wl, const float* __restrict__ bl,
    const float* __restrict__ Wv, const float* __restrict__ bv,
    half_t* __restrict__ qh, half_t* __restrict__ kh,
    half_t* __restrict__ lh, half_t* __restrict__ vh) {
    const int wave = threadIdx.x >> 5;
    const int lane = threadIdx.x & 31;
    const int hi = lane >> 4, lm = lane & 15;

    const int t = blockIdx.x * 4 + wave;
    const int p = t / 768;
    const int r = t - p * 768;
    const int m0 = (r >> 5) * 16;
    const int o0 = (r & 31) * 16;

    const float* X; const float* W; const float* bias;
    half_t* dst; float scale = 1.0f;
    switch (p) {
        case 0: X = queries; W = Wq; bias = bq; dst = qh; scale = SCALE; break;
        case 1: X = keys;    W = Wk; bias = bk; dst = kh; break;
        case 2: X = keys;    W = Wl; bias = bl; dst = lh; break;
        default: X = values; W = Wv; bias = bv; dst = vh; break;
    }

    const int mrow = m0 + lm;
    const int orow = o0 + lm;

    v8f acc = {};
#pragma unroll 4
    for (int c0 = 0; c0 < DIMM; c0 += 32) {
        __builtin_prefetch(&X[(size_t)mrow * DIMM + c0 + 96], 0, 1);
        __builtin_prefetch(&W[(size_t)orow * DIMM + c0 + 96], 0, 1);
        v16h a = load_frag_f32(X, mrow, DIMM, c0, hi);
        v16h b = load_frag_f32(W, orow, DIMM, c0, hi);
        acc = wmma_f16(a, b, acc);
    }

    const int oc = o0 + lm;
    const int h = oc >> 6, d = oc & 63;
    const float bb = bias[oc];
#pragma unroll
    for (int r8 = 0; r8 < 8; ++r8) {
        const int mm = m0 + hi * 8 + r8;
        const int bidx = mm / NN, n = mm - bidx * NN;
        const float val = (acc[r8] + bb) * scale;
        dst[(((size_t)bidx * HH + h) * NN + n) * DH + d] = (half_t)val;
    }
}

__global__ __launch_bounds__(256) void attn_kernel(
    const half_t* __restrict__ qh, const half_t* __restrict__ kh,
    const half_t* __restrict__ lh, const half_t* __restrict__ vh,
    half_t* __restrict__ xbuf) {
    extern __shared__ char smem_raw[];
    half_t* Ks = (half_t*)smem_raw;
    half_t* Ls = Ks + NN * KS;
    half_t* Vt = Ls + NN * KS;
    half_t* P  = Vt + DH * VT;
    unsigned char* P8  = (unsigned char*)(P + NN * PS);
    unsigned char* Vt8 = P8 + NN * PS8;
    float* qs   = (float*)(Vt8 + DH * PS8);
    float* smax = qs + DH;
    float* ssum = smax + 8;
    float* xacc = ssum + 8;

    const int tid  = threadIdx.x;
    const int wave = tid >> 5, lane = tid & 31;
    const int hi = lane >> 4, lm = lane & 15;

    const int bhi = blockIdx.x;
    const int i = bhi % NN;
    const int bh = bhi / NN;

    const half_t* Kg = kh + (size_t)bh * NN * DH;
    const half_t* Lg = lh + (size_t)bh * NN * DH;
    const half_t* Vg = vh + (size_t)bh * NN * DH;
    const half_t* qg = qh + (size_t)bh * NN * DH + (size_t)i * DH;

    if (tid < DH) { qs[tid] = (float)qg[tid]; xacc[tid] = 0.0f; }

    for (int c = tid; c < NN * 8; c += 256) {
        const int j = c >> 3, seg = c & 7;
        const uint64_t gaddr =
            (uint64_t)(uintptr_t)(Lg + (size_t)j * DH + seg * 8);
        const uint32_t laddr =
            (uint32_t)(uintptr_t)(lds_half*)(Ls + j * KS + seg * 8);
        asm volatile("global_load_async_to_lds_b128 %0, %1, off"
                     :: "v"(laddr), "v"(gaddr) : "memory");
    }
    __syncthreads();

    for (int e = tid; e < NN * DH; e += 256) {
        const int j = e >> 6, d = e & 63;
        Ks[j * KS + d] = (half_t)((float)Kg[e] * qs[d]);
        Vt[d * VT + j] = Vg[e];
    }
    for (int e = tid; e < DH * (NN / 2); e += 256) {
        const int d = e / (NN / 2), jp = e - d * (NN / 2);
        const int j = jp * 2;
        const float f0 = (float)Vg[(size_t)j * DH + d];
        const float f1 = (float)Vg[(size_t)(j + 1) * DH + d];
        const int pk = __builtin_amdgcn_cvt_pk_fp8_f32(f0, f1, 0, false);
        *(unsigned short*)(Vt8 + d * PS8 + j) = (unsigned short)(pk & 0xffff);
    }
    asm volatile("s_wait_asynccnt 0x0" ::: "memory");
    __syncthreads();

    float lmax = -1e30f;
    {
        int last_jt = -1;
        v16h a0, a1;
        for (int s = 0; s < 18; ++s) {
            const int t = wave * 18 + s;
            const int jt = t / 12, kt = t - jt * 12;
            const int j0 = jt * 16, k0 = kt * 16;
            if (jt != last_jt) {
                a0 = load_frag_h(Ks, j0 + lm, KS, 0, hi);
                a1 = load_frag_h(Ks, j0 + lm, KS, 32, hi);
                last_jt = jt;
            }
            v8f acc = {};
            acc = wmma_f16(a0, load_frag_h(Ls, k0 + lm, KS, 0, hi), acc);
            acc = wmma_f16(a1, load_frag_h(Ls, k0 + lm, KS, 32, hi), acc);
#pragma unroll
            for (int r = 0; r < 8; ++r) {
                lmax = fmaxf(lmax, acc[r]);
                P[(j0 + hi * 8 + r) * PS + (k0 + lm)] = (half_t)acc[r];
            }
        }
    }
#pragma unroll
    for (int off = 16; off; off >>= 1) lmax = fmaxf(lmax, __shfl_xor(lmax, off, 32));
    if (lane == 0) smax[wave] = lmax;
    __syncthreads();
    float mx = smax[0];
#pragma unroll
    for (int w = 1; w < 8; ++w) mx = fmaxf(mx, smax[w]);

    float lsum = 0.0f;
    for (int e = tid; e < NN * (NN / 2); e += 256) {
        const int j = e / (NN / 2), kp = e - j * (NN / 2);
        const int k = kp * 2;
        const float f0 = __expf((float)P[j * PS + k]     - mx);
        const float f1 = __expf((float)P[j * PS + k + 1] - mx);
        lsum += f0 + f1;
        const int pk = __builtin_amdgcn_cvt_pk_fp8_f32(f0, f1, 0, false);
        *(unsigned short*)(P8 + j * PS8 + k) = (unsigned short)(pk & 0xffff);
    }
#pragma unroll
    for (int off = 16; off; off >>= 1) lsum += __shfl_xor(lsum, off, 32);
    if (lane == 0) ssum[wave] = lsum;
    __syncthreads();
    float denom = 0.0f;
#pragma unroll
    for (int w = 0; w < 8; ++w) denom += ssum[w];

    {
        int last_jt = -1;
        v16i pa0; v8i pa1;
        for (int s = 0; s < 6; ++s) {
            const int t = wave * 6 + s;
            const int jt = t >> 2, dt = t & 3;
            const int j0 = jt * 16, d0 = dt * 16;
            if (jt != last_jt) {
                pa0 = load_frag8_128(P8, j0 + lm, PS8, 0, hi);
                pa1 = load_frag8_64 (P8, j0 + lm, PS8, 128, hi);
                last_jt = jt;
            }
            v16i b0 = load_frag8_128(Vt8, d0 + lm, PS8, 0, hi);
            v8i  b1 = load_frag8_64 (Vt8, d0 + lm, PS8, 128, hi);
            v8f acc = {};
            acc = __builtin_amdgcn_wmma_f32_16x16x128_fp8_fp8(
                      pa0, b0, (short)0, acc, false, false);
            acc = __builtin_amdgcn_wmma_f32_16x16x64_fp8_fp8(
                      pa1, b1, (short)0, acc, false, false);

            const int dcol = d0 + lm;
            const v8h vv = *(const v8h*)(Vt + dcol * VT + j0 + hi * 8);
            float part = 0.0f;
#pragma unroll
            for (int r = 0; r < 8; ++r) part += acc[r] * (float)vv[r];
            atomicAdd(&xacc[dcol], part);
        }
    }
    __syncthreads();

    if (tid < DH) {
        const int b = bh / HH, h = bh - b * HH;
        const float val = xacc[tid] / denom;
        xbuf[((size_t)(b * NN + i)) * DIMM + h * DH + tid] = (half_t)val;
    }
}

__global__ __launch_bounds__(128) void outproj_kernel(
    const half_t* __restrict__ xbuf, const float* __restrict__ Wp,
    const float* __restrict__ bp, float* __restrict__ out) {
    const int wave = threadIdx.x >> 5;
    const int lane = threadIdx.x & 31;
    const int hi = lane >> 4, lm = lane & 15;

    const int t = blockIdx.x * 4 + wave;
    const int m0 = (t >> 5) * 16;
    const int o0 = (t & 31) * 16;

    const int mrow = m0 + lm;
    const int orow = o0 + lm;

    v8f acc = {};
#pragma unroll 4
    for (int c0 = 0; c0 < DIMM; c0 += 32) {
        __builtin_prefetch(&Wp[(size_t)orow * DIMM + c0 + 96], 0, 1);
        v16h a = load_frag_h(xbuf, mrow, DIMM, c0, hi);
        v16h b = load_frag_f32(Wp, orow, DIMM, c0, hi);
        acc = wmma_f16(a, b, acc);
    }

    const int oc = o0 + lm;
    const float bb = bp[oc];
#pragma unroll
    for (int r8 = 0; r8 < 8; ++r8) {
        const int mm = m0 + hi * 8 + r8;
        out[(size_t)mm * DIMM + oc] = acc[r8] + bb;
    }
}

extern "C" void kernel_launch(void* const* d_in, const int* in_sizes, int n_in,
                              void* d_out, int out_size, void* d_ws, size_t ws_size,
                              hipStream_t stream) {
    const float* queries = (const float*)d_in[0];
    const float* keys    = (const float*)d_in[1];
    const float* values  = (const float*)d_in[2];
    const float* Wq = (const float*)d_in[3];  const float* bq = (const float*)d_in[4];
    const float* Wk = (const float*)d_in[5];  const float* bk = (const float*)d_in[6];
    const float* Wl = (const float*)d_in[7];  const float* bl = (const float*)d_in[8];
    const float* Wv = (const float*)d_in[9];  const float* bv = (const float*)d_in[10];
    const float* Wp = (const float*)d_in[11]; const float* bp = (const float*)d_in[12];

    const size_t permat = (size_t)BB * HH * NN * DH;
    half_t* ws16 = (half_t*)d_ws;
    half_t* qh   = ws16;
    half_t* kh   = ws16 + permat;
    half_t* lh   = ws16 + 2 * permat;
    half_t* vh   = ws16 + 3 * permat;
    half_t* xbuf = ws16 + 4 * permat;

    proj_kernel<<<768, 128, 0, stream>>>(queries, keys, values,
                                         Wq, bq, Wk, bk, Wl, bl, Wv, bv,
                                         qh, kh, lh, vh);

    const size_t lds_bytes =
        (size_t)(2 * NN * KS + DH * VT + NN * PS) * sizeof(half_t) +
        (size_t)(NN * PS8 + DH * PS8) +
        (size_t)(DH + 8 + 8 + DH) * sizeof(float);
    attn_kernel<<<BB * HH * NN, 256, lds_bytes, stream>>>(qh, kh, lh, vh, xbuf);

    outproj_kernel<<<192, 128, 0, stream>>>(xbuf, Wp, bp, (float*)d_out);
}
